// LSTM_autoreg_torchscript_38293928411395
// MI455X (gfx1250) — matrix operand order repaired and checked by reading, not compiled, not run on hardware
//
#include <hip/hip_runtime.h>

// ---------------------------------------------------------------------------
// Problem constants
// ---------------------------------------------------------------------------
#define B_     4096
#define NLEV   60
#define LCHUNK 15       // levels per LSTM2 chunk (4 chunks)
#define NX     15
#define NXSFC  24
#define NY     5
#define H_     192
#define NHMEM  16
#define G_     768      // 4*H
#define K1     224      // [x(31) | pad | h(192)]
#define A1_STRIDE 232   // bf16 elems; bank-conflict-free, rows 16B aligned
#define AH_STRIDE 200   // [h(192)] + pad
#define XG_STRIDE 776   // staged xg2 tile (bf16) + pad
#define W1_STRIDE 40    // LDS Wih1 rows (32 cols + pad)
#define GSTRIDE   772   // f32 gate tile + pad
#define NT     4        // N-tiles per wave (12 waves x 4 x 16 = 768 cols)
#define NWAVE  12
#define NTHR   384      // 12 waves -> 3 waves/SIMD -> ~341 VGPR budget
#define CELLS  8        // 16*192 / 384 state cells per thread

typedef __bf16 bf16;
typedef __attribute__((ext_vector_type(16))) __bf16 v16bf;
typedef __attribute__((ext_vector_type(8)))  __bf16 bf16x8;
typedef __attribute__((ext_vector_type(8)))  float  v8f;

union ABfrag { v16bf v; bf16x8 h[2]; };

__device__ __forceinline__ float sigmf(float x) { return 1.0f / (1.0f + __expf(-x)); }

// ---------------------------------------------------------------------------
// Kernel 0: bf16 weight packing.
// Wih1x [768][32] (cols 0..30 = Wih1, col 31 = 0), Whh1c/Wih2c/Whh2c [768][192]
// ---------------------------------------------------------------------------
__global__ __launch_bounds__(256)
void prep_weights(const float* __restrict__ Wih1, const float* __restrict__ Whh1,
                  const float* __restrict__ Wih2, const float* __restrict__ Whh2,
                  bf16* __restrict__ Wih1x, bf16* __restrict__ Whh1c,
                  bf16* __restrict__ Wih2c, bf16* __restrict__ Whh2c)
{
    int idx = blockIdx.x * 256 + threadIdx.x;
    if (idx < G_ * 32) {
        int g = idx >> 5, c = idx & 31;
        Wih1x[idx] = (bf16)((c < 31) ? Wih1[g * 31 + c] : 0.0f);
    }
    if (idx < G_ * H_) {
        Whh1c[idx] = (bf16)Whh1[idx];
        Wih2c[idx] = (bf16)Wih2[idx];
        Whh2c[idx] = (bf16)Whh2[idx];
    }
}

// ---------------------------------------------------------------------------
// Kernel 1: LSTM1, levels 59 -> 0.  384 threads, 16 batch rows per WG.
// Whh1 register-resident (24 frags = 192 VGPRs/wave); Wih1 staged in LDS.
// ---------------------------------------------------------------------------
__global__ __launch_bounds__(NTHR)
void lstm1_kernel(const float* __restrict__ x_lev, const float* __restrict__ x_sfc,
                  const float* __restrict__ h_mem,
                  const float* __restrict__ xmean_lev, const float* __restrict__ xdiv_lev,
                  const float* __restrict__ xmean_sca, const float* __restrict__ xdiv_sca,
                  const float* __restrict__ Wsfc1, const float* __restrict__ bsfc1,
                  const float* __restrict__ Wsfc2, const float* __restrict__ bsfc2,
                  const float* __restrict__ b1,
                  const bf16* __restrict__ Wih1x, const bf16* __restrict__ Whh1c,
                  bf16* __restrict__ hs1)
{
    __shared__ bf16  A1[16 * A1_STRIDE];       // [x(32) | h(192)] rows, bf16
    __shared__ bf16  W1L[G_ * W1_STRIDE];      // Wih1 (768 x 32) in LDS
    __shared__ float gates[16 * GSTRIDE];
    __shared__ float snl[16 * NXSFC];

    const int tid  = threadIdx.x;
    const int b0   = blockIdx.x * 16;
    const int lane = tid & 31;
    const int wave = tid >> 5;                 // 0..11
    const int half = lane >> 4;
    const int mw   = lane & 15;
    const int g0w  = wave * 64;                // 64 gate cols per wave

    // ---- stage Wih1 into LDS (padded rows), surface vector, pad column ----
    {
        const unsigned int* src = (const unsigned int*)Wih1x;   // 12288 dwords
        unsigned int* dst = (unsigned int*)W1L;
        #pragma unroll 1
        for (int r = 0; r < 32; ++r) {
            int idx = tid + NTHR * r;
            int g = idx >> 4, kd = idx & 15;
            dst[g * (W1_STRIDE / 2) + kd] = src[idx];
        }
    }
    {
        int m = tid / NXSFC, i = tid % NXSFC;  // 384 == 16*24 exactly
        snl[tid] = (x_sfc[(size_t)(b0 + m) * NXSFC + i] - xmean_sca[i]) / xdiv_sca[i];
    }
    if (tid < 16) A1[tid * A1_STRIDE + 31] = (bf16)0.0f;
    __syncthreads();

    // ---- h0/c0 ----
    float creg[CELLS];
    #pragma unroll
    for (int r = 0; r < CELLS; ++r) {
        int idx = tid + NTHR * r;              // 0..3071
        int m = idx / H_, k = idx % H_;
        float a0 = bsfc1[k], a1 = bsfc2[k];
        #pragma unroll 4
        for (int i = 0; i < NXSFC; ++i) {
            float s = snl[m * NXSFC + i];
            a0 += s * Wsfc1[k * NXSFC + i];
            a1 += s * Wsfc2[k * NXSFC + i];
        }
        A1[m * A1_STRIDE + 32 + k] = (bf16)tanhf(a0);
        creg[r] = tanhf(a1);
    }

    // ---- preload Whh1 fragments + bias (after init: short live-range start) --
    ABfrag bw[6][NT];
    float  bias[NT];
    #pragma unroll
    for (int nt = 0; nt < NT; ++nt) {
        bias[nt] = b1[g0w + nt * 16 + mw];
        #pragma unroll
        for (int kk = 0; kk < 6; ++kk) {
            const bf16* bp = &Whh1c[(size_t)(g0w + nt * 16 + mw) * H_ + kk * 32 + 16 * half];
            bw[kk][nt].h[0] = *(const bf16x8*)(bp);
            bw[kk][nt].h[1] = *(const bf16x8*)(bp + 8);
        }
    }
    __syncthreads();

    // ---- scan: original level 59 down to 0 ----
    for (int l = NLEV - 1; l >= 0; --l) {
        // fill x-part (16 rows x 31 cols; col 31 stays zero)
        #pragma unroll 1
        for (int q = 0; q < 2; ++q) {
            int idx = tid + NTHR * q;
            if (idx < 512) {
                int m = idx >> 5, c = idx & 31;
                if (c < 31) {
                    float v;
                    if (c < NX) {
                        int o = l * NX + c;
                        v = (x_lev[((size_t)(b0 + m) * NLEV + l) * NX + c] - xmean_lev[o]) / xdiv_lev[o];
                    } else {
                        v = h_mem[((size_t)(b0 + m) * NLEV + l) * NHMEM + (c - NX)];
                    }
                    A1[m * A1_STRIDE + c] = (bf16)v;
                }
            }
        }
        __syncthreads();

        // gate GEMM: kk'=0 uses LDS Wih1, kk'=1..6 use resident Whh1
        v8f acc[NT];
        #pragma unroll
        for (int nt = 0; nt < NT; ++nt) {
            #pragma unroll
            for (int r = 0; r < 8; ++r) acc[nt][r] = bias[nt];
        }
        {
            ABfrag a;
            const bf16* ap = &A1[mw * A1_STRIDE + 16 * half];
            a.h[0] = *(const bf16x8*)(ap);
            a.h[1] = *(const bf16x8*)(ap + 8);
            #pragma unroll
            for (int nt = 0; nt < NT; ++nt) {
                ABfrag bx;
                const bf16* bp = &W1L[(g0w + nt * 16 + mw) * W1_STRIDE + 16 * half];
                bx.h[0] = *(const bf16x8*)(bp);
                bx.h[1] = *(const bf16x8*)(bp + 8);
                acc[nt] = __builtin_amdgcn_wmma_f32_16x16x32_bf16(
                    false, a.v, false, bx.v, (short)0, acc[nt], false, false);
            }
        }
        #pragma unroll
        for (int kk = 0; kk < 6; ++kk) {
            ABfrag a;
            const bf16* ap = &A1[mw * A1_STRIDE + 32 + kk * 32 + 16 * half];
            a.h[0] = *(const bf16x8*)(ap);
            a.h[1] = *(const bf16x8*)(ap + 8);
            #pragma unroll
            for (int nt = 0; nt < NT; ++nt) {
                acc[nt] = __builtin_amdgcn_wmma_f32_16x16x32_bf16(
                    false, a.v, false, bw[kk][nt].v, (short)0, acc[nt], false, false);
            }
        }
        #pragma unroll
        for (int nt = 0; nt < NT; ++nt) {
            #pragma unroll
            for (int r = 0; r < 8; ++r)
                gates[(r + 8 * half) * GSTRIDE + g0w + nt * 16 + mw] = acc[nt][r];
        }
        __syncthreads();

        // cell update
        #pragma unroll
        for (int r = 0; r < CELLS; ++r) {
            int idx = tid + NTHR * r;
            int m = idx / H_, k = idx % H_;
            const float* gm = &gates[m * GSTRIDE];
            float gi = gm[k], gf = gm[H_ + k], gg = gm[2 * H_ + k], go = gm[3 * H_ + k];
            float c = sigmf(gf) * creg[r] + sigmf(gi) * tanhf(gg);
            creg[r] = c;
            bf16 hb = (bf16)(sigmf(go) * tanhf(c));
            A1[m * A1_STRIDE + 32 + k] = hb;
            hs1[((size_t)(b0 + m) * NLEV + l) * H_ + k] = hb;
        }
        __syncthreads();
    }
}

// ---------------------------------------------------------------------------
// Kernel 2: xg2 = hs1 @ Wih2^T + b2 for one 15-level chunk (dense GEMM,
// Wih2 register-resident).  xg2 layout: [lt][b][g] (level-major, bf16).
// ---------------------------------------------------------------------------
__global__ __launch_bounds__(NTHR)
void xg2_gemm(const bf16* __restrict__ Wih2c, const float* __restrict__ b2,
              const bf16* __restrict__ hs1, bf16* __restrict__ xg2, int l0)
{
    const int tid  = threadIdx.x;
    const int lane = tid & 31;
    const int wave = tid >> 5;
    const int half = lane >> 4;
    const int mw   = lane & 15;
    const int g0w  = wave * 64;

    ABfrag bw[6][NT];
    float  bias[NT];
    #pragma unroll
    for (int nt = 0; nt < NT; ++nt) {
        bias[nt] = b2[g0w + nt * 16 + mw];
        #pragma unroll
        for (int kk = 0; kk < 6; ++kk) {
            const bf16* bp = &Wih2c[(size_t)(g0w + nt * 16 + mw) * H_ + kk * 32 + 16 * half];
            bw[kk][nt].h[0] = *(const bf16x8*)(bp);
            bw[kk][nt].h[1] = *(const bf16x8*)(bp + 8);
        }
    }

    #pragma unroll 1
    for (int t = 0; t < 8; ++t) {              // 8 M-tiles per WG
        int row0 = (blockIdx.x * 8 + t) * 16;  // flat row = lt*4096 + b
        int lt = row0 >> 12;
        int b  = (row0 & 4095) + mw;
        const bf16* arow = &hs1[((size_t)b * NLEV + (l0 + lt)) * H_];
        if (t + 1 < 8) __builtin_prefetch(arow + 16 * NLEV * H_, 0, 1);

        v8f acc[NT];
        #pragma unroll
        for (int nt = 0; nt < NT; ++nt) {
            #pragma unroll
            for (int r = 0; r < 8; ++r) acc[nt][r] = bias[nt];
        }
        #pragma unroll
        for (int kk = 0; kk < 6; ++kk) {
            ABfrag a;
            const bf16* ap = arow + kk * 32 + 16 * half;
            a.h[0] = *(const bf16x8*)(ap);
            a.h[1] = *(const bf16x8*)(ap + 8);
            #pragma unroll
            for (int nt = 0; nt < NT; ++nt) {
                acc[nt] = __builtin_amdgcn_wmma_f32_16x16x32_bf16(
                    false, a.v, false, bw[kk][nt].v, (short)0, acc[nt], false, false);
            }
        }
        #pragma unroll
        for (int nt = 0; nt < NT; ++nt) {
            #pragma unroll
            for (int r = 0; r < 8; ++r)
                xg2[(size_t)(row0 + r + 8 * half) * G_ + g0w + nt * 16 + mw] = (bf16)acc[nt][r];
        }
    }
}

// ---------------------------------------------------------------------------
// Kernel 3: LSTM2 scan over one 15-level chunk + heads.
// Only Whh2 resident (24 frags); xg2 tile staged from global each step.
// h/c state carried between chunks via hstate/cstate.
// ---------------------------------------------------------------------------
__global__ __launch_bounds__(NTHR)
void lstm2_kernel(const float* __restrict__ x_sfc,
                  const float* __restrict__ xmean_sca, const float* __restrict__ xdiv_sca,
                  const float* __restrict__ Wtoa1, const float* __restrict__ btoa1,
                  const float* __restrict__ Wtoa2, const float* __restrict__ btoa2,
                  const float* __restrict__ Wlat, const float* __restrict__ blat,
                  const float* __restrict__ Wout, const float* __restrict__ bout,
                  const float* __restrict__ Wsfcout, const float* __restrict__ bsfcout,
                  const float* __restrict__ yscale_lev, const float* __restrict__ yscale_sca,
                  const bf16* __restrict__ Whh2c, const bf16* __restrict__ xg2,
                  float* __restrict__ hstate, float* __restrict__ cstate,
                  float* __restrict__ out, float* __restrict__ out_sfc,
                  float* __restrict__ lat_out, int l0, int first, int last)
{
    __shared__ bf16  A2h[16 * AH_STRIDE];      // h2 rows, bf16
    __shared__ bf16  xgb[16 * XG_STRIDE];      // staged xg2 tile, bf16
    __shared__ float gates[16 * GSTRIDE];
    __shared__ float hbuf[16 * H_];            // f32 h2 for heads/state
    __shared__ float latb[16 * 16];

    const int tid  = threadIdx.x;
    const int b0   = blockIdx.x * 16;
    const int lane = tid & 31;
    const int wave = tid >> 5;
    const int half = lane >> 4;
    const int mw   = lane & 15;
    const int g0w  = wave * 64;

    // ---- init / restore h,c ----
    float creg[CELLS];
    #pragma unroll
    for (int r = 0; r < CELLS; ++r) {
        int idx = tid + NTHR * r;
        int m = idx / H_, k = idx % H_;
        float h;
        if (first) {
            float t0 = (x_sfc[(size_t)(b0 + m) * NXSFC + 0] - xmean_sca[0]) / xdiv_sca[0];
            float t1 = (x_sfc[(size_t)(b0 + m) * NXSFC + 1] - xmean_sca[1]) / xdiv_sca[1];
            h = tanhf(t0 * Wtoa1[k * 2] + t1 * Wtoa1[k * 2 + 1] + btoa1[k]);
            creg[r] = tanhf(t0 * Wtoa2[k * 2] + t1 * Wtoa2[k * 2 + 1] + btoa2[k]);
        } else {
            h = hstate[(size_t)(b0 + m) * H_ + k];
            creg[r] = cstate[(size_t)(b0 + m) * H_ + k];
        }
        A2h[m * AH_STRIDE + k] = (bf16)h;
    }

    // ---- preload Whh2 fragments ----
    ABfrag bw[6][NT];
    #pragma unroll
    for (int nt = 0; nt < NT; ++nt) {
        #pragma unroll
        for (int kk = 0; kk < 6; ++kk) {
            const bf16* bp = &Whh2c[(size_t)(g0w + nt * 16 + mw) * H_ + kk * 32 + 16 * half];
            bw[kk][nt].h[0] = *(const bf16x8*)(bp);
            bw[kk][nt].h[1] = *(const bf16x8*)(bp + 8);
        }
    }
    __syncthreads();

    const unsigned int* xg2w = (const unsigned int*)xg2;   // 2 bf16 per dword

    for (int lt = 0; lt < LCHUNK; ++lt) {
        const int l = l0 + lt;
        // stage xg2 tile [16 x 768] (6144 dwords)
        #pragma unroll 1
        for (int r = 0; r < 16; ++r) {
            int idx = tid + NTHR * r;          // dword index 0..6143
            int m = idx / (G_ / 2), kd = idx % (G_ / 2);
            unsigned int v = xg2w[((size_t)lt * B_ + b0 + m) * (G_ / 2) + kd];
            *(unsigned int*)&xgb[m * XG_STRIDE + kd * 2] = v;
        }
        if (lt + 1 < LCHUNK) {                 // gfx1250 global_prefetch_b8
            int idx = tid * 16;
            int m = idx / (G_ / 2), kd = idx % (G_ / 2);
            __builtin_prefetch(&xg2w[((size_t)(lt + 1) * B_ + b0 + m) * (G_ / 2) + kd], 0, 1);
        }
        __syncthreads();

        // acc init from staged xg2 (already includes b2), then recurrent GEMM
        v8f acc[NT];
        #pragma unroll
        for (int nt = 0; nt < NT; ++nt) {
            #pragma unroll
            for (int r = 0; r < 8; ++r)
                acc[nt][r] = (float)xgb[(r + 8 * half) * XG_STRIDE + g0w + nt * 16 + mw];
        }
        #pragma unroll
        for (int kk = 0; kk < 6; ++kk) {
            ABfrag a;
            const bf16* ap = &A2h[mw * AH_STRIDE + kk * 32 + 16 * half];
            a.h[0] = *(const bf16x8*)(ap);
            a.h[1] = *(const bf16x8*)(ap + 8);
            #pragma unroll
            for (int nt = 0; nt < NT; ++nt) {
                acc[nt] = __builtin_amdgcn_wmma_f32_16x16x32_bf16(
                    false, a.v, false, bw[kk][nt].v, (short)0, acc[nt], false, false);
            }
        }
        #pragma unroll
        for (int nt = 0; nt < NT; ++nt) {
            #pragma unroll
            for (int r = 0; r < 8; ++r)
                gates[(r + 8 * half) * GSTRIDE + g0w + nt * 16 + mw] = acc[nt][r];
        }
        __syncthreads();

        // cell update
        #pragma unroll
        for (int r = 0; r < CELLS; ++r) {
            int idx = tid + NTHR * r;
            int m = idx / H_, k = idx % H_;
            const float* gm = &gates[m * GSTRIDE];
            float gi = gm[k], gf = gm[H_ + k], gg = gm[2 * H_ + k], go = gm[3 * H_ + k];
            float c = sigmf(gf) * creg[r] + sigmf(gi) * tanhf(gg);
            creg[r] = c;
            float h = sigmf(go) * tanhf(c);
            A2h[m * AH_STRIDE + k] = (bf16)h;
            hbuf[m * H_ + k] = h;
        }
        __syncthreads();

        // lat head (threads 0..255: one [m,j] entry each)
        if (tid < 256) {
            int m = tid >> 4, j = tid & 15;
            float a = blat[j];
            #pragma unroll 4
            for (int k = 0; k < H_; ++k) a += hbuf[m * H_ + k] * Wlat[j * H_ + k];
            latb[m * 16 + j] = a;
            lat_out[((size_t)(b0 + m) * NLEV + l) * NHMEM + j] = a;
        }
        __syncthreads();

        // out head
        if (tid < 80) {
            int m = tid / 5, y = tid % 5;
            float a = bout[y];
            #pragma unroll
            for (int j = 0; j < 16; ++j) a += latb[m * 16 + j] * Wout[y * 16 + j];
            out[((size_t)(b0 + m) * NLEV + l) * NY + y] = a / yscale_lev[l * NY + y];
        }
        __syncthreads();
    }

    if (!last) {
        // save carried state
        #pragma unroll
        for (int r = 0; r < CELLS; ++r) {
            int idx = tid + NTHR * r;
            int m = idx / H_, k = idx % H_;
            hstate[(size_t)(b0 + m) * H_ + k] = hbuf[m * H_ + k];
            cstate[(size_t)(b0 + m) * H_ + k] = creg[r];
        }
    } else if (tid < 80) {
        // out_sfc from h_last
        int m = tid / 5, y = tid % 5;
        float a = bsfcout[y];
        #pragma unroll 4
        for (int k = 0; k < H_; ++k) a += hbuf[m * H_ + k] * Wsfcout[y * H_ + k];
        out_sfc[(size_t)(b0 + m) * NY + y] = a / yscale_sca[y];
    }
}

// ---------------------------------------------------------------------------
// Launch
// ---------------------------------------------------------------------------
extern "C" void kernel_launch(void* const* d_in, const int* in_sizes, int n_in,
                              void* d_out, int out_size, void* d_ws, size_t ws_size,
                              hipStream_t stream)
{
    const float* x_lev     = (const float*)d_in[0];
    const float* x_sfc     = (const float*)d_in[1];
    const float* h_mem     = (const float*)d_in[2];
    const float* xmean_lev = (const float*)d_in[3];
    const float* xdiv_lev  = (const float*)d_in[4];
    const float* xmean_sca = (const float*)d_in[5];
    const float* xdiv_sca  = (const float*)d_in[6];
    const float* yscale_lev= (const float*)d_in[7];
    const float* yscale_sca= (const float*)d_in[8];
    const float* Wih1      = (const float*)d_in[9];
    const float* Whh1      = (const float*)d_in[10];
    const float* b1        = (const float*)d_in[11];
    const float* Wih2      = (const float*)d_in[12];
    const float* Whh2      = (const float*)d_in[13];
    const float* b2        = (const float*)d_in[14];
    const float* Wsfc1     = (const float*)d_in[15];
    const float* bsfc1     = (const float*)d_in[16];
    const float* Wsfc2     = (const float*)d_in[17];
    const float* bsfc2     = (const float*)d_in[18];
    const float* Wtoa1     = (const float*)d_in[19];
    const float* btoa1     = (const float*)d_in[20];
    const float* Wtoa2     = (const float*)d_in[21];
    const float* btoa2     = (const float*)d_in[22];
    const float* Wlat      = (const float*)d_in[23];
    const float* blat      = (const float*)d_in[24];
    const float* Wout      = (const float*)d_in[25];
    const float* bout      = (const float*)d_in[26];
    const float* Wsfcout   = (const float*)d_in[27];
    const float* bsfcout   = (const float*)d_in[28];

    // workspace layout (bytes, 16B-aligned blocks)
    char* ws = (char*)d_ws;
    bf16*  Wih1x  = (bf16*)(ws);                       //   49,152
    bf16*  Whh1c  = (bf16*)(ws + 49152);               //  294,912
    bf16*  Wih2c  = (bf16*)(ws + 344064);              //  294,912
    bf16*  Whh2c  = (bf16*)(ws + 638976);              //  294,912
    bf16*  hs1    = (bf16*)(ws + 933888);              // 94,371,840
    bf16*  xg2    = (bf16*)(ws + 95305728);            // 94,371,840 (per-chunk, reused)
    float* hstate = (float*)(ws + 189677568);          //  3,145,728
    float* cstate = (float*)(ws + 192823296);          //  3,145,728  (total ~196 MB)

    float* out     = (float*)d_out;                    // [B,60,5]
    float* out_sfc = out + (size_t)B_ * NLEV * NY;     // [B,5]
    float* lat_out = out_sfc + (size_t)B_ * NY;        // [B,60,16]

    prep_weights<<<(G_ * H_ + 255) / 256, 256, 0, stream>>>(
        Wih1, Whh1, Wih2, Whh2, Wih1x, Whh1c, Wih2c, Whh2c);

    lstm1_kernel<<<B_ / 16, NTHR, 0, stream>>>(
        x_lev, x_sfc, h_mem, xmean_lev, xdiv_lev, xmean_sca, xdiv_sca,
        Wsfc1, bsfc1, Wsfc2, bsfc2, b1, Wih1x, Whh1c, hs1);

    for (int c = 0; c < NLEV / LCHUNK; ++c) {
        int l0 = c * LCHUNK;
        xg2_gemm<<<(LCHUNK * B_ / 16) / 8, NTHR, 0, stream>>>(Wih2c, b2, hs1, xg2, l0);
        lstm2_kernel<<<B_ / 16, NTHR, 0, stream>>>(
            x_sfc, xmean_sca, xdiv_sca, Wtoa1, btoa1, Wtoa2, btoa2,
            Wlat, blat, Wout, bout, Wsfcout, bsfcout, yscale_lev, yscale_sca,
            Whh2c, xg2, hstate, cstate, out, out_sfc, lat_out,
            l0, c == 0 ? 1 : 0, c == (NLEV / LCHUNK - 1) ? 1 : 0);
    }
}
